// GATNet_41721312313874
// MI455X (gfx1250) — compile-verified, matrix operand order checked by reading, not run on hardware
//
#include <hip/hip_runtime.h>
#include <hip/hip_bf16.h>

// ---------------------------------------------------------------------------
// GATNet (EGAT + 3xGATConv) inference for MI455X (gfx1250, wave32, WMMA).
// Dense transforms use v_wmma_f32_16x16x32_bf16 (bf16 in, fp32 accum).
// GEMM is templated on K so all bounds checks fold; A rows are loaded with
// float4 (global_load_b128). All row counts are multiples of 16, so no
// per-lane row guards are needed (EXEC stays all-ones around WMMA).
// Workspace requirement: ~130M floats (~520 MB), dominated by E x 64 edge buf.
// ---------------------------------------------------------------------------

#define N_NODES   100000
#define N_EDGES   1600000
#define N_GRAPHS  512
#define HIDDEN    64
#define HEADS     4
#define HEAD_DIM  16
#define IN_FEAT   32
#define EDGE_FEAT 16
#define N_CLASSES 10
#define N_CONV    3
#define BN_EPS    1e-5f
#define BN_SHIFT  1e-4f
#define NEG_SLOPE 0.2f

typedef __attribute__((ext_vector_type(16))) __bf16 v16bf;
typedef __attribute__((ext_vector_type(8)))  float  v8f;

__device__ __forceinline__ __bf16 f2bf(float f) {
  unsigned u = __float_as_uint(f);
  u += 0x7fffu + ((u >> 16) & 1u);          // round-to-nearest-even
  return __builtin_bit_cast(__bf16, (unsigned short)(u >> 16));
}
// monotonic float <-> uint mapping for atomicMax on floats
__device__ __forceinline__ unsigned f2ord(float f) {
  unsigned u = __float_as_uint(f);
  return (u & 0x80000000u) ? ~u : (u | 0x80000000u);
}
__device__ __forceinline__ float ord2f(unsigned o) {
  unsigned u = (o & 0x80000000u) ? (o & 0x7fffffffu) : ~o;
  return __uint_as_float(u);
}
__device__ __forceinline__ float leaky(float x) { return x > 0.f ? x : NEG_SLOPE * x; }

// ---------------------------------------------------------------------------
// WMMA GEMM: C[M,64] = act( ((A - mu)*rs + 1e-4) @ B + bias ),  K in {16,32,64}.
// A: M x K fp32 row-major (M % 16 == 0). B: K x 64 fp32 row-major -> bf16.
// One wave computes a 16x64 strip: 4 f32 accumulators, K in chunks of 32.
// bf16 fragment layout (ISA 7.12.2, 16-bit A 16x32 / symmetric B 32x16):
//   lane<16 : K = kb..kb+7, kb+16..kb+23 ; lane>=16 : K = kb+8..kb+15, kb+24..31
// ---------------------------------------------------------------------------
template <int K, bool BN>
__global__ void __launch_bounds__(256)
gemm_bn_wmma(const float* __restrict__ A, const float* __restrict__ B,
             const float* __restrict__ bias, const float* __restrict__ mu,
             const float* __restrict__ rs, float* __restrict__ C,
             int M, int relu)
{
  const int lane = threadIdx.x & 31;
  const int wv   = threadIdx.x >> 5;
  const int rowBase = (blockIdx.x * 8 + wv) * 16;
  if (rowBase >= M) return;                      // wave-uniform: EXEC stays all-ones

  const int n0  = lane & 15;
  const int kb0 = (lane >> 4) << 3;              // 0 or 8
  constexpr int KC = (K + 31) / 32;              // 1 or 2 K-chunks

  // Preload B fragments (B is tiny: <= 64x64, L2/L0 resident). Guards fold.
  v16bf bfr[KC][4];
  #pragma unroll
  for (int kc = 0; kc < KC; ++kc) {
    #pragma unroll
    for (int t = 0; t < 4; ++t) {
      const int n  = t * 16 + n0;
      const int kb = kc * 32 + kb0;
      #pragma unroll
      for (int j = 0; j < 8; ++j) {
        int k  = kb + j;
        int k2 = kb + 16 + j;
        bfr[kc][t][j]     = (k  < K) ? f2bf(B[k  * 64 + n]) : f2bf(0.f);
        bfr[kc][t][8 + j] = (k2 < K) ? f2bf(B[k2 * 64 + n]) : f2bf(0.f);
      }
    }
  }

  const int r = rowBase + n0;                    // M % 16 == 0 -> always in range
  const float* arow = A + (size_t)r * K;

  v8f acc0 = {}, acc1 = {}, acc2 = {}, acc3 = {};
  #pragma unroll
  for (int kc = 0; kc < KC; ++kc) {
    const int kb = kc * 32 + kb0;
    // lower half: 8 contiguous floats, 16B-aligned for all K in {16,32,64}
    float va[16];
    *(float4*)(va + 0) = *(const float4*)(arow + kb);
    *(float4*)(va + 4) = *(const float4*)(arow + kb + 4);
    if ((kc + 1) * 32 <= K) {                    // compile-time after unroll
      *(float4*)(va + 8)  = *(const float4*)(arow + kb + 16);
      *(float4*)(va + 12) = *(const float4*)(arow + kb + 20);
    } else {
      #pragma unroll
      for (int j = 8; j < 16; ++j) va[j] = 0.f;
    }
    v16bf af;
    #pragma unroll
    for (int j = 0; j < 8; ++j) {
      int k  = kb + j;
      int k2 = kb + 16 + j;
      float v0 = va[j];
      float v1 = va[8 + j];
      if (BN) {                                  // fused BatchNorm on A columns
        v0 = (v0 - mu[k]) * rs[k] + BN_SHIFT;
        if (k2 < K) v1 = (v1 - mu[k2]) * rs[k2] + BN_SHIFT;
      }
      af[j]     = f2bf(v0);
      af[8 + j] = f2bf(v1);
    }
    acc0 = __builtin_amdgcn_wmma_f32_16x16x32_bf16(false, af, false, bfr[kc][0], (short)0, acc0, false, false);
    acc1 = __builtin_amdgcn_wmma_f32_16x16x32_bf16(false, af, false, bfr[kc][1], (short)0, acc1, false, false);
    acc2 = __builtin_amdgcn_wmma_f32_16x16x32_bf16(false, af, false, bfr[kc][2], (short)0, acc2, false, false);
    acc3 = __builtin_amdgcn_wmma_f32_16x16x32_bf16(false, af, false, bfr[kc][3], (short)0, acc3, false, false);
  }

  // C/D layout: VGPR v -> row v (lanes<16) / v+8 (lanes>=16), col = lane&15
  const int rOff = (lane >> 4) * 8;
  float bv[4];
  #pragma unroll
  for (int t = 0; t < 4; ++t) bv[t] = bias ? bias[t * 16 + n0] : 0.f;
  #pragma unroll
  for (int v = 0; v < 8; ++v) {
    const int rr = rowBase + rOff + v;
    float o0 = acc0[v] + bv[0], o1 = acc1[v] + bv[1];
    float o2 = acc2[v] + bv[2], o3 = acc3[v] + bv[3];
    if (relu) { o0 = fmaxf(o0, 0.f); o1 = fmaxf(o1, 0.f); o2 = fmaxf(o2, 0.f); o3 = fmaxf(o3, 0.f); }
    float* cp = C + (size_t)rr * 64 + n0;
    cp[0] = o0; cp[16] = o1; cp[32] = o2; cp[48] = o3;
  }
}

// ---------------------------------------------------------------------------
// Utility kernels
// ---------------------------------------------------------------------------
__global__ void fill_f32(float* __restrict__ p, float v, long n) {
  long i  = (long)blockIdx.x * blockDim.x + threadIdx.x;
  long st = (long)gridDim.x * blockDim.x;
  for (; i < n; i += st) p[i] = v;
}

// column sums + sums of squares (C must be power of two <= 64)
__global__ void colstats(const float* __restrict__ X, long n, int cmask,
                         float* __restrict__ stats) {
  __shared__ float ls[128];
  const int C = cmask + 1;
  if ((int)threadIdx.x < 2 * C) ls[threadIdx.x] = 0.f;
  __syncthreads();
  long i  = (long)blockIdx.x * blockDim.x + threadIdx.x;
  long st = (long)gridDim.x * blockDim.x;
  for (; i < n; i += st) {
    float v = X[i];
    int c = (int)(i & (long)cmask);
    atomicAdd(&ls[c], v);
    atomicAdd(&ls[C + c], v * v);
  }
  __syncthreads();
  if ((int)threadIdx.x < 2 * C) atomicAdd(&stats[threadIdx.x], ls[threadIdx.x]);
}

__global__ void bn_finalize(const float* __restrict__ stats, int C, float invM,
                            float* __restrict__ mu, float* __restrict__ rs) {
  int c = threadIdx.x;
  if (c < C) {
    float m = stats[c] * invM;
    float v = stats[C + c] * invM - m * m;       // biased variance (torch BN train)
    mu[c] = m;
    rs[c] = rsqrtf(fmaxf(v, 0.f) + BN_EPS);
  }
}

__global__ void bias_relu(const float* __restrict__ in, const float* __restrict__ bias,
                          float* __restrict__ out, long n) {
  long i = (long)blockIdx.x * blockDim.x + threadIdx.x;
  if (i < n) {
    float v = in[i] + (bias ? bias[i & 63] : 0.f);
    out[i] = fmaxf(v, 0.f);
  }
}

// ---------------------------------------------------------------------------
// EGAT: wa[k] = (We @ att)[k] (k<16), wa[16] = be . att
// ---------------------------------------------------------------------------
__global__ void egat_fold(const float* __restrict__ We, const float* __restrict__ be,
                          const float* __restrict__ att, float* __restrict__ wa) {
  int t = threadIdx.x;
  if (t < EDGE_FEAT) {
    float s = 0.f;
    for (int j = 0; j < HIDDEN; ++j) s += We[t * HIDDEN + j] * att[j];
    wa[t] = s;
  } else if (t == EDGE_FEAT) {
    float s = 0.f;
    for (int j = 0; j < HIDDEN; ++j) s += be[j] * att[j];
    wa[t] = s;
  }
}

// hs[n] = h[n] . att   (one wave per node, butterfly reduce)
__global__ void node_att1(const float* __restrict__ h, const float* __restrict__ att,
                          float* __restrict__ hs, int n) {
  int lane = threadIdx.x & 31;
  int node = (int)(((long)blockIdx.x * blockDim.x + threadIdx.x) >> 5);
  if (node >= n) return;
  const float* r = h + (size_t)node * 64;
  float p = r[lane] * att[lane] + r[lane + 32] * att[lane + 32];
  #pragma unroll
  for (int m = 16; m; m >>= 1) p += __shfl_xor(p, m, 32);
  if (lane == 0) hs[node] = p;
}

__global__ void egat_logit(const float* __restrict__ ea, const float* __restrict__ wa,
                           const float* __restrict__ hs, const int* __restrict__ ei,
                           float* __restrict__ s, unsigned* __restrict__ mord, int E) {
  int e = blockIdx.x * blockDim.x + threadIdx.x;
  if (e >= E) return;
  const float* row = ea + (size_t)e * EDGE_FEAT;
  float ae = wa[EDGE_FEAT];                      // be . att
  #pragma unroll
  for (int k = 0; k < EDGE_FEAT; ++k) ae += row[k] * wa[k];
  int src = ei[e], dst = ei[E + e];
  float v = leaky(hs[src] + hs[dst] + ae);
  s[e] = v;
  atomicMax(&mord[dst], f2ord(v));
}

__global__ void egat_exp(float* __restrict__ s, const unsigned* __restrict__ mord,
                         const int* __restrict__ ei, float* __restrict__ ssum, int E) {
  int e = blockIdx.x * blockDim.x + threadIdx.x;
  if (e >= E) return;
  int dst = ei[E + e];
  float w = __expf(s[e] - ord2f(mord[dst]));
  s[e] = w;
  atomicAdd(&ssum[dst], w);
}

__global__ void egat_agg(const float* __restrict__ w, const float* __restrict__ ssum,
                         const float* __restrict__ h, const int* __restrict__ ei,
                         float* __restrict__ out, int E) {
  long i = (long)blockIdx.x * blockDim.x + threadIdx.x;
  if (i >= (long)E * 8) return;
  int e = (int)(i >> 3), c0 = ((int)i & 7) * 8;
  int src = ei[e], dst = ei[E + e];
  float alpha = w[e] / ssum[dst];
  const float* hp = h + (size_t)src * 64 + c0;
  float* op = out + (size_t)dst * 64 + c0;
  #pragma unroll
  for (int j = 0; j < 8; ++j) atomicAdd(&op[j], alpha * hp[j]);
}

// ---------------------------------------------------------------------------
// GATConv: waC[h*64+k] = sum_d We[k][h*16+d] * att_edge[h][d]
// ---------------------------------------------------------------------------
__global__ void conv_fold(const float* __restrict__ We, const float* __restrict__ atte,
                          float* __restrict__ waC) {
  int t = threadIdx.x;                           // t = h*64 + k, 256 threads
  int hh = t >> 6, k = t & 63;
  float s = 0.f;
  #pragma unroll
  for (int d = 0; d < HEAD_DIM; ++d) s += We[k * 64 + hh * 16 + d] * atte[hh * 16 + d];
  waC[t] = s;
}

// out[n*4+h] = hh[n, h*16 : h*16+16] . attv[h]
__global__ void node_att4(const float* __restrict__ hh, const float* __restrict__ attv,
                          float* __restrict__ out, int n) {
  long i = (long)blockIdx.x * blockDim.x + threadIdx.x;
  if (i >= (long)n * 4) return;
  int node = (int)(i >> 2), hd = (int)i & 3;
  const float* r = hh + (size_t)node * 64 + hd * 16;
  const float* a = attv + hd * 16;
  float s = 0.f;
  #pragma unroll
  for (int d = 0; d < HEAD_DIM; ++d) s += r[d] * a[d];
  out[i] = s;
}

// one wave per edge: ae[h] = edge[e] . waC[h], logits + atomicMax
__global__ void conv_logit(const float* __restrict__ edgef, const float* __restrict__ waC,
                           const float* __restrict__ pa, const float* __restrict__ pd,
                           const int* __restrict__ ei, float* __restrict__ logit,
                           unsigned* __restrict__ mord, int E) {
  int lane = threadIdx.x & 31;
  int e = (int)(((long)blockIdx.x * blockDim.x + threadIdx.x) >> 5);
  if (e >= E) return;
  const float* row = edgef + (size_t)e * 64;
  float x0 = row[lane], x1 = row[lane + 32];
  float p0 = x0 * waC[lane]        + x1 * waC[lane + 32];
  float p1 = x0 * waC[64 + lane]   + x1 * waC[96 + lane];
  float p2 = x0 * waC[128 + lane]  + x1 * waC[160 + lane];
  float p3 = x0 * waC[192 + lane]  + x1 * waC[224 + lane];
  #pragma unroll
  for (int m = 16; m; m >>= 1) {
    p0 += __shfl_xor(p0, m, 32); p1 += __shfl_xor(p1, m, 32);
    p2 += __shfl_xor(p2, m, 32); p3 += __shfl_xor(p3, m, 32);
  }
  if (lane < 4) {
    float ae = (lane == 0) ? p0 : (lane == 1) ? p1 : (lane == 2) ? p2 : p3;
    int src = ei[e], dst = ei[E + e];
    float v = leaky(pa[src * 4 + lane] + pd[dst * 4 + lane] + ae);
    logit[(size_t)e * 4 + lane] = v;
    atomicMax(&mord[dst * 4 + lane], f2ord(v));
  }
}

__global__ void conv_exp(float* __restrict__ logit, const unsigned* __restrict__ mord,
                         const int* __restrict__ ei, float* __restrict__ ssum, int E) {
  long i = (long)blockIdx.x * blockDim.x + threadIdx.x;
  if (i >= (long)E * 4) return;
  int e = (int)(i >> 2), hd = (int)i & 3;
  int dst = ei[E + e];
  float w = __expf(logit[i] - ord2f(mord[dst * 4 + hd]));
  logit[i] = w;
  atomicAdd(&ssum[dst * 4 + hd], w);
}

__global__ void conv_agg(const float* __restrict__ logit, const float* __restrict__ ssum,
                         const float* __restrict__ hh, const int* __restrict__ ei,
                         float* __restrict__ out, int E) {
  long i = (long)blockIdx.x * blockDim.x + threadIdx.x;
  if (i >= (long)E * 4) return;
  int e = (int)(i >> 2), hd = (int)i & 3;
  int src = ei[e], dst = ei[E + e];
  float alpha = logit[i] / ssum[dst * 4 + hd];
  const float* hp = hh + (size_t)src * 64 + hd * 16;
  float* op = out + (size_t)dst * 64 + hd * 16;
  #pragma unroll
  for (int d = 0; d < HEAD_DIM; ++d) atomicAdd(&op[d], alpha * hp[d]);
}

// ---------------------------------------------------------------------------
// global add pool + head
// ---------------------------------------------------------------------------
__global__ void pool_kernel(const float* __restrict__ x, const int* __restrict__ batch,
                            float* __restrict__ pool, int n) {
  long i = (long)blockIdx.x * blockDim.x + threadIdx.x;
  if (i >= (long)n * 8) return;
  int node = (int)(i >> 3), c0 = ((int)i & 7) * 8;
  int g = batch[node];
  const float* xp = x + (size_t)node * 64 + c0;
  float* pp = pool + (size_t)g * 64 + c0;
  #pragma unroll
  for (int j = 0; j < 8; ++j) atomicAdd(&pp[j], xp[j]);
}

// per-row: BN(y) @ cls_W + cls_b, then log_softmax (one wave per row)
__global__ void cls_logsoftmax(const float* __restrict__ y, const float* __restrict__ mu,
                               const float* __restrict__ rs, const float* __restrict__ W,
                               const float* __restrict__ b, float* __restrict__ out) {
  int lane = threadIdx.x & 31;
  int row = (int)(((long)blockIdx.x * blockDim.x + threadIdx.x) >> 5);
  if (row >= N_GRAPHS) return;
  float logit = -3.0e38f;
  if (lane < N_CLASSES) {
    float acc = b[lane];
    const float* yr = y + (size_t)row * 64;
    #pragma unroll
    for (int k = 0; k < 64; ++k)
      acc += ((yr[k] - mu[k]) * rs[k] + BN_SHIFT) * W[k * N_CLASSES + lane];
    logit = acc;
  }
  float mx = logit;
  #pragma unroll
  for (int m = 16; m; m >>= 1) mx = fmaxf(mx, __shfl_xor(mx, m, 32));
  float ex = (lane < N_CLASSES) ? __expf(logit - mx) : 0.f;
  float se = ex;
  #pragma unroll
  for (int m = 16; m; m >>= 1) se += __shfl_xor(se, m, 32);
  if (lane < N_CLASSES) out[(size_t)row * N_CLASSES + lane] = logit - mx - __logf(se);
}

// ---------------------------------------------------------------------------
// host orchestration
// ---------------------------------------------------------------------------
static inline int nblk(long n, int tb) { return (int)((n + tb - 1) / tb); }
static inline int capblk(long n, int tb, int cap) { int g = nblk(n, tb); return g > cap ? cap : g; }

extern "C" void kernel_launch(void* const* d_in, const int* in_sizes, int n_in,
                              void* d_out, int out_size, void* d_ws, size_t ws_size,
                              hipStream_t stream) {
  (void)in_sizes; (void)n_in; (void)out_size; (void)ws_size;
  const float* x         = (const float*)d_in[0];
  const float* ea        = (const float*)d_in[1];
  const int*   ei        = (const int*)  d_in[2];
  const int*   batch     = (const int*)  d_in[3];
  const float* egat_Wx   = (const float*)d_in[4];
  const float* egat_bx   = (const float*)d_in[5];
  const float* egat_We   = (const float*)d_in[6];
  const float* egat_be   = (const float*)d_in[7];
  const float* egat_att  = (const float*)d_in[8];
  const float* conv_W    = (const float*)d_in[9];
  const float* conv_We   = (const float*)d_in[10];
  const float* att_src   = (const float*)d_in[11];
  const float* att_dst   = (const float*)d_in[12];
  const float* att_edge  = (const float*)d_in[13];
  const float* conv_bias = (const float*)d_in[14];
  const float* lin_W     = (const float*)d_in[15];
  const float* lin_b     = (const float*)d_in[16];
  const float* cls_W     = (const float*)d_in[17];
  const float* cls_b     = (const float*)d_in[18];
  float* out = (float*)d_out;

  float* ws = (float*)d_ws;
  size_t off = 0;
  auto alloc = [&](size_t n) { float* p = ws + off; off += (n + 63) & ~(size_t)63; return p; };
  float*    hbuf  = alloc((size_t)N_NODES * 64);   // h / hh
  float*    edgef = alloc((size_t)N_EDGES * 64);   // relu(he), fixed across convs
  float*    xbuf  = alloc((size_t)N_NODES * 64);   // current node features
  float*    obuf  = alloc((size_t)N_NODES * 64);   // scatter accumulator
  float*    pa    = alloc((size_t)N_NODES * 4);    // src attention terms (EGAT uses [0..N))
  float*    pd    = alloc((size_t)N_NODES * 4);    // dst attention terms
  unsigned* mord  = (unsigned*)alloc((size_t)N_NODES * 4);
  float*    ssum  = alloc((size_t)N_NODES * 4);
  float*    wbuf  = alloc((size_t)N_EDGES * 4);    // logits -> softmax weights
  float*    pool  = alloc((size_t)N_GRAPHS * 64);
  float*    ybuf  = alloc((size_t)N_GRAPHS * 64);
  float*    stats = alloc(128);
  float*    muv   = alloc(64);
  float*    rsv   = alloc(64);
  float*    wa    = alloc(320);

  const int TB = 256;

  // ---- BN over x, fused into h GEMM ----
  fill_f32<<<1, 128, 0, stream>>>(stats, 0.f, 128);
  colstats<<<capblk((long)N_NODES * 32, TB, 4096), TB, 0, stream>>>(x, (long)N_NODES * 32, 31, stats);
  bn_finalize<<<1, 64, 0, stream>>>(stats, 32, 1.f / N_NODES, muv, rsv);
  gemm_bn_wmma<32, true><<<nblk(N_NODES, 128), TB, 0, stream>>>(
      x, egat_Wx, egat_bx, muv, rsv, hbuf, N_NODES, 0);

  // ---- edge = relu(edge_attr @ We + be) ----
  gemm_bn_wmma<16, false><<<nblk(N_EDGES, 128), TB, 0, stream>>>(
      ea, egat_We, egat_be, nullptr, nullptr, edgef, N_EDGES, 1);

  // ---- EGAT attention + aggregation ----
  egat_fold<<<1, 32, 0, stream>>>(egat_We, egat_be, egat_att, wa);
  node_att1<<<nblk((long)N_NODES * 32, TB), TB, 0, stream>>>(hbuf, egat_att, pa, N_NODES);
  fill_f32<<<nblk(N_NODES, TB), TB, 0, stream>>>((float*)mord, 0.f, N_NODES);   // ord(min)
  fill_f32<<<nblk(N_NODES, TB), TB, 0, stream>>>(ssum, 0.f, N_NODES);
  fill_f32<<<nblk((long)N_NODES * 64, TB), TB, 0, stream>>>(obuf, 0.f, (long)N_NODES * 64);
  egat_logit<<<nblk(N_EDGES, TB), TB, 0, stream>>>(ea, wa, pa, ei, wbuf, mord, N_EDGES);
  egat_exp<<<nblk(N_EDGES, TB), TB, 0, stream>>>(wbuf, mord, ei, ssum, N_EDGES);
  egat_agg<<<nblk((long)N_EDGES * 8, TB), TB, 0, stream>>>(wbuf, ssum, hbuf, ei, obuf, N_EDGES);
  bias_relu<<<nblk((long)N_NODES * 64, TB), TB, 0, stream>>>(obuf, nullptr, xbuf, (long)N_NODES * 64);

  // ---- 3 x GATConv ----
  for (int i = 0; i < N_CONV; ++i) {
    fill_f32<<<1, 128, 0, stream>>>(stats, 0.f, 128);
    colstats<<<capblk((long)N_NODES * 64, TB, 4096), TB, 0, stream>>>(xbuf, (long)N_NODES * 64, 63, stats);
    bn_finalize<<<1, 64, 0, stream>>>(stats, 64, 1.f / N_NODES, muv, rsv);
    gemm_bn_wmma<64, true><<<nblk(N_NODES, 128), TB, 0, stream>>>(
        xbuf, conv_W + (size_t)i * 64 * 64, nullptr, muv, rsv, hbuf, N_NODES, 0);
    conv_fold<<<1, 256, 0, stream>>>(conv_We + (size_t)i * 64 * 64, att_edge + (size_t)i * 64, wa);
    node_att4<<<nblk((long)N_NODES * 4, TB), TB, 0, stream>>>(hbuf, att_src + (size_t)i * 64, pa, N_NODES);
    node_att4<<<nblk((long)N_NODES * 4, TB), TB, 0, stream>>>(hbuf, att_dst + (size_t)i * 64, pd, N_NODES);
    fill_f32<<<nblk((long)N_NODES * 4, TB), TB, 0, stream>>>((float*)mord, 0.f, (long)N_NODES * 4);
    fill_f32<<<nblk((long)N_NODES * 4, TB), TB, 0, stream>>>(ssum, 0.f, (long)N_NODES * 4);
    fill_f32<<<nblk((long)N_NODES * 64, TB), TB, 0, stream>>>(obuf, 0.f, (long)N_NODES * 64);
    conv_logit<<<nblk((long)N_EDGES * 32, TB), TB, 0, stream>>>(edgef, wa, pa, pd, ei, wbuf, mord, N_EDGES);
    conv_exp<<<nblk((long)N_EDGES * 4, TB), TB, 0, stream>>>(wbuf, mord, ei, ssum, N_EDGES);
    conv_agg<<<nblk((long)N_EDGES * 4, TB), TB, 0, stream>>>(wbuf, ssum, hbuf, ei, obuf, N_EDGES);
    bias_relu<<<nblk((long)N_NODES * 64, TB), TB, 0, stream>>>(obuf, conv_bias + (size_t)i * 64,
                                                               xbuf, (long)N_NODES * 64);
  }

  // ---- global add pool ----
  fill_f32<<<nblk((long)N_GRAPHS * 64, TB), TB, 0, stream>>>(pool, 0.f, (long)N_GRAPHS * 64);
  pool_kernel<<<nblk((long)N_NODES * 8, TB), TB, 0, stream>>>(xbuf, batch, pool, N_NODES);

  // ---- BN + lin + relu ----
  fill_f32<<<1, 128, 0, stream>>>(stats, 0.f, 128);
  colstats<<<nblk((long)N_GRAPHS * 64, TB), TB, 0, stream>>>(pool, (long)N_GRAPHS * 64, 63, stats);
  bn_finalize<<<1, 64, 0, stream>>>(stats, 64, 1.f / N_GRAPHS, muv, rsv);
  gemm_bn_wmma<64, true><<<nblk(N_GRAPHS, 128), TB, 0, stream>>>(
      pool, lin_W, lin_b, muv, rsv, ybuf, N_GRAPHS, 1);

  // ---- BN + classifier + log_softmax ----
  fill_f32<<<1, 128, 0, stream>>>(stats, 0.f, 128);
  colstats<<<nblk((long)N_GRAPHS * 64, TB), TB, 0, stream>>>(ybuf, (long)N_GRAPHS * 64, 63, stats);
  bn_finalize<<<1, 64, 0, stream>>>(stats, 64, 1.f / N_GRAPHS, muv, rsv);
  cls_logsoftmax<<<nblk((long)N_GRAPHS * 32, TB), TB, 0, stream>>>(ybuf, muv, rsv, cls_W, cls_b, out);
}
